// DS_ODE_2465311228271
// MI455X (gfx1250) — compile-verified, matrix operand order checked by reading, not run on hardware
//
#include <hip/hip_runtime.h>
#include <math.h>

// ---------------- problem constants ----------------
#define BB     32
#define NN     5
#define TT     20
#define SS     4
#define PPC    2
#define PMM    8
#define HH     64
#define KKC    3
#define PREDN  40
#define LLN    6
#define HEADSN 4
#define DD     1280          // H*T
#define DFFN   2560
#define DHN    320           // D/HEADS
#define SUBN   10
#define KBN    96            // K*B
#define ROWS   480           // KB*N
#define ROWS2  9600          // ROWS*T

#define MCH    240           // M-chunk per block (ROWS/2)
#define MT     15            // WMMA M-tiles per chunk
#define BN     128           // N panel per block (8 waves x 16)

// ---------------- WMMA types ----------------
typedef __attribute__((ext_vector_type(16))) __bf16  v16bf;
typedef __attribute__((ext_vector_type(8)))  __bf16  v8bf;
typedef __attribute__((ext_vector_type(8)))  float   v8f;

__device__ __forceinline__ v16bf mkfrag(const __bf16* lo, const __bf16* hi) {
  v8bf a = *reinterpret_cast<const v8bf*>(lo);
  v8bf b = *reinterpret_cast<const v8bf*>(hi);
  v16bf c;
#pragma unroll
  for (int i = 0; i < 8; ++i) { c[i] = a[i]; c[i + 8] = b[i]; }
  return c;
}

// ================= encoder: pin -> h1 -> z0 (written to z_t[t=0]) ==========
__global__ void k_encoder(const float* __restrict__ data, const float* __restrict__ phy,
                          const float* __restrict__ Wi1, const float* __restrict__ bi1,
                          const float* __restrict__ Wi2, const float* __restrict__ bi2,
                          float* __restrict__ zt) {
  int row = blockIdx.x;                // (k*B+b)*N + n
  int j   = threadIdx.x;               // 0..63
  int kb  = row / NN, n = row % NN;
  int k   = kb / BB,  b = kb % BB;
  __shared__ float pin[14];
  __shared__ float h1[HH];
  if (j < SS)                      pin[j] = data[((b*NN + n)*TT + 0)*SS + j];
  else if (j < SS + PPC)           pin[j] = data[((b*NN + n)*TT + 0)*SS + (j - SS)];
  else if (j < SS + PPC + PMM)     pin[j] = phy[b*PMM + (j - SS - PPC)];
  __syncthreads();
  float a = bi1[k*HH + j];
  for (int i = 0; i < 14; ++i) a += pin[i] * Wi1[(k*14 + i)*HH + j];
  h1[j] = tanhf(a);
  __syncthreads();
  float z = bi2[k*HH + j];
  for (int i = 0; i < HH; ++i) z += h1[i] * Wi2[(k*HH + i)*HH + j];
  zt[(row*TT + 0)*HH + j] = z;
}

// ================= RK4 ODE scan: 480 independent rows, 1 row / 64-thr block =
// Weight columns live in VGPRs (w1c[68], w2c[64]); shared cat/hb vectors are
// read as b128 broadcasts; 2-wave barriers keep the serial chain tight.
__global__ void __launch_bounds__(64) k_ode(const float* __restrict__ data,
    const float* __restrict__ Wo1, const float* __restrict__ bo1,
    const float* __restrict__ Wo2, const float* __restrict__ bo2,
    float* __restrict__ zt, float* __restrict__ lastz) {
  int j   = threadIdx.x;               // 0..63 output dim
  int row = blockIdx.x;                // 0..479
  int kb  = row / NN, n = row % NN, b = kb % BB;
  __shared__ __align__(16) float cat[HH + SS];
  __shared__ __align__(16) float hb[HH];
  __shared__ float obsr[TT * SS];
  float w1c[HH + SS], w2c[HH];
#pragma unroll
  for (int i = 0; i < HH + SS; ++i) w1c[i] = Wo1[i*HH + j];
#pragma unroll
  for (int i = 0; i < HH; ++i)      w2c[i] = Wo2[i*HH + j];
  float b1 = bo1[j], b2 = bo2[j];
  for (int i = j; i < TT * SS; i += 64)
    obsr[i] = data[(b*NN + n)*TT*SS + i];
  __syncthreads();
  float zj = zt[(row*TT + 0)*HH + j];
  const float h = 0.1f;

  auto f = [&](float tv, float ztmp) -> float {
    cat[j] = ztmp;
    if (j < SS) {
      int i0 = (int)floorf(tv);
      i0 = i0 < 0 ? 0 : (i0 > TT - 2 ? TT - 2 : i0);
      float w = tv - (float)i0;
      cat[HH + j] = (1.0f - w) * obsr[i0*SS + j] + w * obsr[(i0 + 1)*SS + j];
    }
    __syncthreads();
    float a = b1;
    const float4* cp = reinterpret_cast<const float4*>(cat);
#pragma unroll
    for (int q = 0; q < (HH + SS) / 4; ++q) {
      float4 c = cp[q];
      a += c.x*w1c[4*q] + c.y*w1c[4*q+1] + c.z*w1c[4*q+2] + c.w*w1c[4*q+3];
    }
    hb[j] = tanhf(a);
    __syncthreads();
    float o = b2;
    const float4* hp = reinterpret_cast<const float4*>(hb);
#pragma unroll
    for (int q = 0; q < HH / 4; ++q) {
      float4 hv = hp[q];
      o += hv.x*w2c[4*q] + hv.y*w2c[4*q+1] + hv.z*w2c[4*q+2] + hv.w*w2c[4*q+3];
    }
    return o;
  };

  for (int step = 0; step < (TT - 1) * SUBN; ++step) {
    float t0 = (float)step * h;
    float k1 = f(t0,          zj);
    float k2 = f(t0 + 0.5f*h, zj + 0.5f*h*k1);
    float k3 = f(t0 + 0.5f*h, zj + 0.5f*h*k2);
    float k4 = f(t0 + h,      zj + h*k3);
    zj += (h / 6.0f) * (k1 + 2.0f*k2 + 2.0f*k3 + k4);
    if (((step + 1) % SUBN) == 0)
      zt[(row*TT + (step + 1) / SUBN)*HH + j] = zj;
  }
  lastz[row*HH + j] = zj;
}

// ================= mu = relu(obs@We1+be1)@We2+be2 ===========================
__global__ void k_mu(const float* __restrict__ data,
                     const float* __restrict__ We1, const float* __restrict__ be1,
                     const float* __restrict__ We2, const float* __restrict__ be2,
                     float* __restrict__ mu, float* __restrict__ lastmu) {
  int tid = threadIdx.x;
  int g = tid >> 6, j = tid & 63;
  int row2 = blockIdx.x * 4 + g;         // < 9600
  int t  = row2 % TT;
  int rn = row2 / TT;                    // kb*N + n
  int kb = rn / NN, n = rn % NN, b = kb % BB;
  __shared__ float sh[4][HH];
  float s0[SS];
  for (int i = 0; i < SS; ++i) s0[i] = data[((b*NN + n)*TT + t)*SS + i];
  float a = be1[j];
  for (int i = 0; i < SS; ++i) a += s0[i] * We1[i*HH + j];
  sh[g][j] = fmaxf(a, 0.0f);
  __syncthreads();
  float m = be2[j];
  for (int i = 0; i < HH; ++i) m += sh[g][i] * We2[i*HH + j];
  mu[row2*HH + j] = m;
  if (t == TT - 1) lastmu[rn*HH + j] = m;
}

// ================= proj + gate -> x (480 x 1280) ============================
__global__ void k_projgate(const float* __restrict__ zt, const float* __restrict__ mu,
                           const float* __restrict__ Wf, const float* __restrict__ bf_,
                           const float* __restrict__ Wg1, const float* __restrict__ bg1,
                           const float* __restrict__ Wg2, const float* __restrict__ bg2,
                           float* __restrict__ x) {
  int tid = threadIdx.x;
  int g = tid >> 6, j = tid & 63;
  int row2 = blockIdx.x * 4 + g;
  int t  = row2 % TT;
  int rn = row2 / TT;
  int k  = rn / (BB * NN);
  __shared__ float c4[4][2 * HH];
  __shared__ float p4[4][HH];
  __shared__ float gh4[4][HH];
  c4[g][j]      = zt[row2*HH + j];
  c4[g][HH + j] = mu[row2*HH + j];
  __syncthreads();
  float p = bf_[j];
  for (int i = 0; i < 2 * HH; ++i) p += c4[g][i] * Wf[i*HH + j];
  p4[g][j] = p;
  __syncthreads();
  float a = bg1[k*HH + j];
  for (int i = 0; i < HH; ++i) a += p4[g][i] * Wg1[(k*HH + i)*HH + j];
  gh4[g][j] = tanhf(a);
  __syncthreads();
  float s = bg2[k*HH + j];
  for (int i = 0; i < HH; ++i) s += gh4[g][i] * Wg2[(k*HH + i)*HH + j];
  float gate = 1.0f / (1.0f + expf(-s));
  x[(long)rn*DD + t*HH + j] = p4[g][j] * gate;
}

// ================= bf16 WMMA GEMM (software pipelined) ======================
// C = act(A@W + bias [+ Res]).  A: (480,Kd) f32, W: (Kd,Nd) f32, C: (480,Nd).
// 8 waves x 16-wide N subtile = 128-wide N panel; block loops over 15 M-tiles
// re-using the staged W tile (weights stream gridDim.y=2 times total).
// Next tile's global loads are issued into registers BEFORE the WMMA block so
// HBM latency overlaps the 15 chained WMMAs.
struct GemmJob { const float* W; const float* bias; float* C; };

template <bool GUARD, bool RELU, bool HASRES>
__global__ void __launch_bounds__(256, 1) k_gemm(
    const float* __restrict__ A, GemmJob j0, GemmJob j1, GemmJob j2,
    const float* __restrict__ Res, int Kd, int Nd) {
  GemmJob job = (blockIdx.z == 0) ? j0 : ((blockIdx.z == 1) ? j1 : j2);
  const float* __restrict__ W = job.W;
  __shared__ __align__(16) __bf16 la[MCH][32];   // A chunk tile (240 x 32)
  __shared__ __align__(16) __bf16 lb[BN][32];    // W tile transposed (N,K)
  int tid  = (int)threadIdx.x;
  int n0   = (int)blockIdx.x * BN;
  int m0   = (int)blockIdx.y * MCH;
  int wid  = tid >> 5, lane = tid & 31;
  int half = lane >> 4, l = lane & 15;
  int gk   = half * 8;                 // K-group base per ISA 16-bit A layout
  int wn   = wid * 16;                 // wave's N subtile in panel
  v8f acc[MT] = {};
  float4 aR[8];                        // thread's A row (32 floats)
  float4 bR[4];

  auto loadT = [&](int k0) {
    if (tid < MCH) {
#pragma unroll
      for (int r = 0; r < 8; ++r)
        aR[r] = *reinterpret_cast<const float4*>(A + (long)(m0 + tid)*Kd + k0 + r*4);
    }
#pragma unroll
    for (int r = 0; r < 4; ++r) {
      int idx = tid + r * 256;
      int kk  = idx >> 5;              // 0..31
      int nq  = (idx & 31) * 4;        // 0..124
      int n   = n0 + nq;
      if (!GUARD || n < Nd)
        bR[r] = *reinterpret_cast<const float4*>(W + (long)(k0 + kk)*Nd + n);
      else
        bR[r] = make_float4(0.0f, 0.0f, 0.0f, 0.0f);
    }
  };
  auto storeT = [&]() {
    if (tid < MCH) {
#pragma unroll
      for (int rr = 0; rr < 4; ++rr) {
        v8bf p;
        p[0] = (__bf16)aR[2*rr].x;   p[1] = (__bf16)aR[2*rr].y;
        p[2] = (__bf16)aR[2*rr].z;   p[3] = (__bf16)aR[2*rr].w;
        p[4] = (__bf16)aR[2*rr+1].x; p[5] = (__bf16)aR[2*rr+1].y;
        p[6] = (__bf16)aR[2*rr+1].z; p[7] = (__bf16)aR[2*rr+1].w;
        *reinterpret_cast<v8bf*>(&la[tid][rr*8]) = p;
      }
    }
#pragma unroll
    for (int r = 0; r < 4; ++r) {
      int idx = tid + r * 256;
      int kk  = idx >> 5;
      int nq  = (idx & 31) * 4;
      lb[nq+0][kk] = (__bf16)bR[r].x; lb[nq+1][kk] = (__bf16)bR[r].y;
      lb[nq+2][kk] = (__bf16)bR[r].z; lb[nq+3][kk] = (__bf16)bR[r].w;
    }
  };

  loadT(0);
  storeT();
  __syncthreads();
  for (int k0 = 0; k0 < Kd; k0 += 32) {
    bool more = (k0 + 32) < Kd;
    if (more) loadT(k0 + 32);          // prefetch next tile into registers
    v16bf bfrag = mkfrag(&lb[wn + l][gk], &lb[wn + l][gk + 16]);
#pragma unroll
    for (int mt = 0; mt < MT; ++mt) {
      v16bf afrag = mkfrag(&la[mt*16 + l][gk], &la[mt*16 + l][gk + 16]);
      acc[mt] = __builtin_amdgcn_wmma_f32_16x16x32_bf16(
          false, afrag, false, bfrag, (short)0, acc[mt], false, false);
    }
    if (!more) break;
    __syncthreads();                   // all LDS reads of current tile done
    storeT();                          // convert + stage next tile
    __syncthreads();                   // staged tile visible
  }
  // epilogue: VGPR r -> M = r (lanes 0-15) / r+8 (lanes 16-31); N = lane&15
  int n = n0 + wn + l;
  if (GUARD && n >= Nd) return;
  float bn = job.bias[n];
#pragma unroll
  for (int mt = 0; mt < MT; ++mt) {
#pragma unroll
    for (int r = 0; r < 8; ++r) {
      int m = m0 + mt*16 + r + half*8;
      float v = acc[mt][r] + bn;
      if (HASRES) v += Res[(long)m*Nd + n];
      if (RELU)   v = fmaxf(v, 0.0f);
      job.C[(long)m*Nd + n] = v;
    }
  }
}

// ================= attention (N=5 tokens, 4 heads of 320) ===================
__global__ void k_attn_softmax(const float* __restrict__ Q, const float* __restrict__ Km,
                               float* __restrict__ att) {
  int idx = blockIdx.x * blockDim.x + threadIdx.x;
  if (idx >= KBN * HEADSN * NN) return;
  int kb = idx / (HEADSN * NN);
  int h  = (idx / NN) % HEADSN;
  int qi = idx % NN;
  const float scale = 1.0f / sqrtf((float)DHN);
  float sv[NN];
  float mx = -1e30f;
  const float* qp = Q + ((long)(kb*NN + qi))*DD + h*DHN;
  for (int j = 0; j < NN; ++j) {
    const float* kp = Km + ((long)(kb*NN + j))*DD + h*DHN;
    float d = 0.0f;
    for (int c = 0; c < DHN; ++c) d += qp[c] * kp[c];
    sv[j] = d * scale;
    mx = fmaxf(mx, sv[j]);
  }
  float sum = 0.0f;
  for (int j = 0; j < NN; ++j) { sv[j] = expf(sv[j] - mx); sum += sv[j]; }
  float inv = 1.0f / sum;
  for (int j = 0; j < NN; ++j)
    att[((kb*HEADSN + h)*NN + qi)*NN + j] = sv[j] * inv;
}

__global__ void k_attn_av(const float* __restrict__ att, const float* __restrict__ V,
                          float* __restrict__ O) {
  int gid = blockIdx.x * blockDim.x + threadIdx.x;
  if (gid >= ROWS * DD) return;
  int rn = gid / DD, c = gid % DD;
  int kb = rn / NN, qi = rn % NN;
  int h  = c / DHN;
  const float* ap = att + ((kb*HEADSN + h)*NN + qi)*NN;
  float o = 0.0f;
  for (int j = 0; j < NN; ++j) o += ap[j] * V[(long)(kb*NN + j)*DD + c];
  O[gid] = o;
}

// ================= LayerNorm over D=1280 ====================================
__global__ void __launch_bounds__(256) k_ln(const float* __restrict__ in,
    const float* __restrict__ sc, const float* __restrict__ bi,
    float* __restrict__ out) {
  __shared__ float red[256];
  int row = blockIdx.x, tid = threadIdx.x;
  const float* p = in + (long)row * DD;
  float s = 0.0f;
  for (int i = tid; i < DD; i += 256) s += p[i];
  red[tid] = s; __syncthreads();
  for (int o = 128; o > 0; o >>= 1) { if (tid < o) red[tid] += red[tid + o]; __syncthreads(); }
  float mean = red[0] / (float)DD;
  __syncthreads();
  float v = 0.0f;
  for (int i = tid; i < DD; i += 256) { float d = p[i] - mean; v += d * d; }
  red[tid] = v; __syncthreads();
  for (int o = 128; o > 0; o >>= 1) { if (tid < o) red[tid] += red[tid + o]; __syncthreads(); }
  float inv = rsqrtf(red[0] / (float)DD + 1e-5f);
  for (int i = tid; i < DD; i += 256)
    out[(long)row*DD + i] = (p[i] - mean) * inv * sc[i] + bi[i];
}

// ================= host launcher ============================================
extern "C" void kernel_launch(void* const* d_in, const int* in_sizes, int n_in,
                              void* d_out, int out_size, void* d_ws, size_t ws_size,
                              hipStream_t stream) {
  (void)in_sizes; (void)n_in; (void)out_size; (void)ws_size;
  const float* data = (const float*)d_in[0];
  const float* phy  = (const float*)d_in[1];
  const float* Wi1  = (const float*)d_in[2];
  const float* bi1  = (const float*)d_in[3];
  const float* Wi2  = (const float*)d_in[4];
  const float* bi2  = (const float*)d_in[5];
  const float* Wo1  = (const float*)d_in[6];
  const float* bo1  = (const float*)d_in[7];
  const float* Wo2  = (const float*)d_in[8];
  const float* bo2  = (const float*)d_in[9];
  const float* We1  = (const float*)d_in[10];
  const float* be1  = (const float*)d_in[11];
  const float* We2  = (const float*)d_in[12];
  const float* be2  = (const float*)d_in[13];
  const float* Wf   = (const float*)d_in[14];
  const float* bf_  = (const float*)d_in[15];
  const float* Wg1  = (const float*)d_in[16];
  const float* bg1  = (const float*)d_in[17];
  const float* Wg2  = (const float*)d_in[18];
  const float* bg2  = (const float*)d_in[19];
  const float* Wq   = (const float*)d_in[20];
  const float* bq   = (const float*)d_in[21];
  const float* Wk   = (const float*)d_in[22];
  const float* bk   = (const float*)d_in[23];
  const float* Wv   = (const float*)d_in[24];
  const float* bv   = (const float*)d_in[25];
  const float* Woa  = (const float*)d_in[26];
  const float* boa  = (const float*)d_in[27];
  const float* ln1s = (const float*)d_in[28];
  const float* ln1b = (const float*)d_in[29];
  const float* ln2s = (const float*)d_in[30];
  const float* ln2b = (const float*)d_in[31];
  const float* Wff1 = (const float*)d_in[32];
  const float* bff1 = (const float*)d_in[33];
  const float* Wff2 = (const float*)d_in[34];
  const float* bff2 = (const float*)d_in[35];
  const float* Wout = (const float*)d_in[36];
  const float* bout = (const float*)d_in[37];

  float* out    = (float*)d_out;
  float* pred   = out;                          // 76800
  float* lastmu = out + 76800;                  // 30720
  float* lastz  = out + 76800 + 30720;          // 30720

  const long SEG = (long)ROWS * DD;             // 614400 floats
  float* ws  = (float*)d_ws;
  float* zt  = ws;
  float* mu  = ws + 1 * SEG;
  float* x   = ws + 2 * SEG;
  float* Qb  = ws + 3 * SEG;
  float* Kb  = ws + 4 * SEG;
  float* Vb  = ws + 5 * SEG;
  float* Ob  = ws + 6 * SEG;
  float* Buf = ws + 7 * SEG;
  float* Ffh = ws + 8 * SEG;                    // 480*2560
  float* Att = ws + 8 * SEG + (long)ROWS * DFFN;

  k_encoder <<<ROWS,     64,  0, stream>>>(data, phy, Wi1, bi1, Wi2, bi2, zt);
  k_ode     <<<ROWS,     64,  0, stream>>>(data, Wo1, bo1, Wo2, bo2, zt, lastz);
  k_mu      <<<ROWS2/4,  256, 0, stream>>>(data, We1, be1, We2, be2, mu, lastmu);
  k_projgate<<<ROWS2/4,  256, 0, stream>>>(zt, mu, Wf, bf_, Wg1, bg1, Wg2, bg2, x);

  dim3 gQKV(DD / BN,   ROWS / MCH, 3);          // (10, 2, 3)
  dim3 gD  (DD / BN,   ROWS / MCH, 1);          // (10, 2)
  dim3 gF  (DFFN / BN, ROWS / MCH, 1);          // (20, 2)
  dim3 gO  ((PREDN*SS + BN - 1) / BN, ROWS / MCH, 1); // (2, 2)

  for (int l = 0; l < LLN; ++l) {
    GemmJob jq{Wq + (long)l*DD*DD, bq + l*DD, Qb};
    GemmJob jk{Wk + (long)l*DD*DD, bk + l*DD, Kb};
    GemmJob jv{Wv + (long)l*DD*DD, bv + l*DD, Vb};
    k_gemm<false,false,false><<<gQKV, 256, 0, stream>>>(x, jq, jk, jv, nullptr, DD, DD);
    k_attn_softmax<<<(KBN*HEADSN*NN + 255)/256, 256, 0, stream>>>(Qb, Kb, Att);
    k_attn_av     <<<(ROWS*DD + 255)/256,       256, 0, stream>>>(Att, Vb, Ob);
    GemmJob jo{Woa + (long)l*DD*DD, boa + l*DD, Buf};
    k_gemm<false,false,true><<<gD, 256, 0, stream>>>(Ob, jo, jo, jo, x, DD, DD);
    k_ln<<<ROWS, 256, 0, stream>>>(Buf, ln1s + l*DD, ln1b + l*DD, x);
    GemmJob jf1{Wff1 + (long)l*DD*DFFN, bff1 + l*DFFN, Ffh};
    k_gemm<false,true,false><<<gF, 256, 0, stream>>>(x, jf1, jf1, jf1, nullptr, DD, DFFN);
    GemmJob jf2{Wff2 + (long)l*DFFN*DD, bff2 + l*DD, Buf};
    k_gemm<false,false,true><<<gD, 256, 0, stream>>>(Ffh, jf2, jf2, jf2, x, DFFN, DD);
    k_ln<<<ROWS, 256, 0, stream>>>(Buf, ln2s + l*DD, ln2b + l*DD, x);
  }
  GemmJob jout{Wout, bout, pred};
  k_gemm<true,false,false><<<gO, 256, 0, stream>>>(x, jout, jout, jout, nullptr, DD, PREDN*SS);
}
